// RBNN_35613868818872
// MI455X (gfx1250) — matrix operand order repaired and checked by reading, not compiled, not run on hardware
//
#include <hip/hip_runtime.h>
#include <hip/hip_bf16.h>

#define HID 256
#define DT 0.001f
#define ALPHA 0.5f
#define NSTEPS 5

typedef __attribute__((ext_vector_type(16))) __bf16 v16bf;
typedef __attribute__((ext_vector_type(8)))  __bf16 v8bf;
typedef __attribute__((ext_vector_type(8)))  float  v8f;

static __device__ __forceinline__ __bf16 f2bf(float f) {
  union { float f; unsigned int u; } a; a.f = f;
  unsigned int u = a.u;
  u += 0x7fffu + ((u >> 16) & 1u);          // round-to-nearest-even
  unsigned short s = (unsigned short)(u >> 16);
  union { unsigned short s; __bf16 b; } r; r.s = s;
  return r.b;
}

static __device__ __forceinline__ v16bf load_frag(const __bf16* p0, const __bf16* p1) {
  union { v16bf v; v8bf h[2]; } u;
  u.h[0] = *(const v8bf*)p0;
  u.h[1] = *(const v8bf*)p1;
  return u.v;
}

static __device__ __forceinline__ void inv3(const float m[9], float o[9]) {
  float A  = m[4]*m[8] - m[5]*m[7];
  float Bc = m[5]*m[6] - m[3]*m[8];
  float C  = m[3]*m[7] - m[4]*m[6];
  float det = m[0]*A + m[1]*Bc + m[2]*C;
  float id = 1.f / det;
  o[0] = A * id;
  o[1] = (m[2]*m[7] - m[1]*m[8]) * id;
  o[2] = (m[1]*m[5] - m[2]*m[4]) * id;
  o[3] = Bc * id;
  o[4] = (m[0]*m[8] - m[2]*m[6]) * id;
  o[5] = (m[2]*m[3] - m[0]*m[5]) * id;
  o[6] = C * id;
  o[7] = (m[1]*m[6] - m[0]*m[7]) * id;
  o[8] = (m[0]*m[4] - m[1]*m[3]) * id;
}

// ---------------------------------------------------------------------------
// K0: convert W2 -> bf16 (row-major) and W2^T -> bf16 (row-major)
// ---------------------------------------------------------------------------
__global__ void k_convert_w2(const float* __restrict__ W2,
                             __bf16* __restrict__ W2bf,
                             __bf16* __restrict__ W2tbf) {
  int idx = blockIdx.x * 256 + threadIdx.x;   // 65536 elements
  int r = idx >> 8, c = idx & 255;
  __bf16 v = f2bf(W2[idx]);
  W2bf[idx] = v;
  W2tbf[c * 256 + r] = v;
}

// ---------------------------------------------------------------------------
// K1/K3: M = vee(dV^T R - R^T dV) with dV = grad_q V(q), q = R.flatten()
// One workgroup (256 thr = 8 waves) handles 16 bodies; WMMA bf16 GEMMs for
// the 256x256 layer (forward + transpose backward). Each wave computes two
// 16-row tiles against a shared B fragment (loaded once per K step).
// ---------------------------------------------------------------------------
__global__ __launch_bounds__(256) void k_explicitM(
    const float* __restrict__ Rin,
    const float* __restrict__ W1, const float* __restrict__ b1,
    const __bf16* __restrict__ W2bf, const __bf16* __restrict__ W2tbf,
    const float* __restrict__ b2, const float* __restrict__ W3,
    float* __restrict__ Mout)
{
  __shared__ __align__(16) __bf16 h1bf[16][264];   // h1 (relu) bf16, [body][k]
  __shared__ __align__(16) __bf16 gzbf[16][264];   // W3 * relu'(z2) bf16
  __shared__ __align__(16) float  h1f[16][264];    // h1 f32, reused as dz1
  __shared__ float red[16][16][9];                 // dq partial sums
  __shared__ float b2s[256];                       // staged bias / head weights
  __shared__ float W3s[256];

  const int tid = threadIdx.x;
  const int b0 = blockIdx.x * 16;

  // stage b2 / W3 once (avoids predicated global loads in the epilogues)
  b2s[tid] = b2[tid];
  W3s[tid] = W3[tid];

  // ---- layer 1 forward: h1 = relu(W1 q + b1), 16 threads per body ----
  {
    const int body = tid >> 4;
    const int seg  = tid & 15;
    float q[9];
    #pragma unroll
    for (int i = 0; i < 9; ++i) q[i] = Rin[(b0 + body) * 9 + i];
    #pragma unroll
    for (int jj = 0; jj < 16; ++jj) {
      int j = seg * 16 + jj;
      float z = b1[j];
      const float* wr = W1 + j * 9;
      #pragma unroll
      for (int i = 0; i < 9; ++i) z = fmaf(wr[i], q[i], z);
      float h = fmaxf(z, 0.f);
      h1f[body][j]  = h;
      h1bf[body][j] = f2bf(h);
    }
  }
  __syncthreads();

  const int lane = tid & 31;
  const int wv   = tid >> 5;
  const int hi   = lane >> 4;   // half-wave select per ISA fragment layout
  const int lr   = lane & 15;
  const int row0 = wv * 16;           // first row tile for this wave
  const int row1 = wv * 16 + 128;     // second row tile (tiles wv and wv+8)

  // ---- layer 2 forward: z2 = W2 h1 + b2 ; gz = W3 * relu'(z2) ----
  {
    v8f c0 = {}, c1 = {};
    const __bf16* arow0 = W2bf + (row0 + lr) * 256 + (hi ? 8 : 0);
    const __bf16* arow1 = W2bf + (row1 + lr) * 256 + (hi ? 8 : 0);
    for (int k0 = 0; k0 < 256; k0 += 32) {
      const __bf16* bp = &h1bf[lr][k0 + (hi ? 8 : 0)];
      v16bf b  = load_frag(bp, bp + 16);         // shared by both row tiles
      v16bf a0 = load_frag(arow0 + k0, arow0 + k0 + 16);
      v16bf a1 = load_frag(arow1 + k0, arow1 + k0 + 16);
      c0 = __builtin_amdgcn_wmma_f32_16x16x32_bf16(false, a0, false, b,
                                                   (short)0, c0, false, false);
      c1 = __builtin_amdgcn_wmma_f32_16x16x32_bf16(false, a1, false, b,
                                                   (short)0, c1, false, false);
    }
    const int rs0 = row0 + hi * 8;
    const int rs1 = row1 + hi * 8;
    v8bf o0, o1;
    #pragma unroll
    for (int r = 0; r < 8; ++r) {
      float z20 = c0[r] + b2s[rs0 + r];
      float z21 = c1[r] + b2s[rs1 + r];
      o0[r] = f2bf(z20 > 0.f ? W3s[rs0 + r] : 0.f);   // -> v_cndmask
      o1[r] = f2bf(z21 > 0.f ? W3s[rs1 + r] : 0.f);
    }
    *(v8bf*)&gzbf[lr][rs0] = o0;
    *(v8bf*)&gzbf[lr][rs1] = o1;
  }
  __syncthreads();

  // ---- layer 2 backward: dh1 = W2^T gz ; dz1 = dh1 * relu'(z1) ----
  {
    v8f c0 = {}, c1 = {};
    const __bf16* arow0 = W2tbf + (row0 + lr) * 256 + (hi ? 8 : 0);
    const __bf16* arow1 = W2tbf + (row1 + lr) * 256 + (hi ? 8 : 0);
    for (int k0 = 0; k0 < 256; k0 += 32) {
      const __bf16* bp = &gzbf[lr][k0 + (hi ? 8 : 0)];
      v16bf b  = load_frag(bp, bp + 16);
      v16bf a0 = load_frag(arow0 + k0, arow0 + k0 + 16);
      v16bf a1 = load_frag(arow1 + k0, arow1 + k0 + 16);
      c0 = __builtin_amdgcn_wmma_f32_16x16x32_bf16(false, a0, false, b,
                                                   (short)0, c0, false, false);
      c1 = __builtin_amdgcn_wmma_f32_16x16x32_bf16(false, a1, false, b,
                                                   (short)0, c1, false, false);
    }
    const int rs0 = row0 + hi * 8;
    const int rs1 = row1 + hi * 8;
    #pragma unroll
    for (int r = 0; r < 8; ++r) {
      float m0 = h1f[lr][rs0 + r];                 // relu(z1) > 0 <=> z1 > 0
      float m1 = h1f[lr][rs1 + r];
      h1f[lr][rs0 + r] = (m0 > 0.f) ? c0[r] : 0.f; // overwrite with dz1
      h1f[lr][rs1 + r] = (m1 > 0.f) ? c1[r] : 0.f;
    }
  }
  __syncthreads();

  // ---- layer 1 backward: dq = W1^T dz1 (partials across 16 thread-groups) --
  {
    const int body = tid & 15;
    const int g = tid >> 4;
    float acc[9];
    #pragma unroll
    for (int j = 0; j < 9; ++j) acc[j] = 0.f;
    for (int kk = 0; kk < 16; ++kk) {
      int k = g * 16 + kk;
      float d = h1f[body][k];
      const float* wr = W1 + k * 9;
      #pragma unroll
      for (int j = 0; j < 9; ++j) acc[j] = fmaf(d, wr[j], acc[j]);
    }
    #pragma unroll
    for (int j = 0; j < 9; ++j) red[body][g][j] = acc[j];
  }
  __syncthreads();

  // ---- finalize dV, M = (SM[2,1], SM[0,2], SM[1,0]) ----
  if (tid < 16) {
    float dV[9];
    #pragma unroll
    for (int j = 0; j < 9; ++j) {
      float s = 0.f;
      for (int g = 0; g < 16; ++g) s += red[tid][g][j];
      dV[j] = s;
    }
    float Rm[9];
    #pragma unroll
    for (int i = 0; i < 9; ++i) Rm[i] = Rin[(b0 + tid) * 9 + i];
    float m0 = 0.f, m1 = 0.f, m2 = 0.f;  // SM = dV^T R - R^T dV
    #pragma unroll
    for (int k = 0; k < 3; ++k) {
      m0 += dV[k*3+2]*Rm[k*3+1] - Rm[k*3+2]*dV[k*3+1];  // SM[2][1]
      m1 += dV[k*3+0]*Rm[k*3+2] - Rm[k*3+0]*dV[k*3+2];  // SM[0][2]
      m2 += dV[k*3+1]*Rm[k*3+0] - Rm[k*3+1]*dV[k*3+0];  // SM[1][0]
    }
    Mout[(b0 + tid)*3 + 0] = m0;
    Mout[(b0 + tid)*3 + 1] = m1;
    Mout[(b0 + tid)*3 + 2] = m2;
  }
}

// ---------------------------------------------------------------------------
// K2: per-body Newton solve + Cayley map; R_next = R @ F
// ---------------------------------------------------------------------------
__global__ void k_newton(const float* __restrict__ Rin, const float* __restrict__ om_in,
                         const float* __restrict__ Iin, const float* __restrict__ M1,
                         float* __restrict__ Rnext, float* __restrict__ Fout,
                         float* __restrict__ pout, int B)
{
  int b = blockIdx.x * blockDim.x + threadIdx.x;
  if (b >= B) return;
  float Im[9];
  #pragma unroll
  for (int i = 0; i < 9; ++i) Im[i] = Iin[i];
  float om0 = om_in[b*3], om1 = om_in[b*3+1], om2 = om_in[b*3+2];
  float p[3];
  #pragma unroll
  for (int i = 0; i < 3; ++i) p[i] = Im[i*3]*om0 + Im[i*3+1]*om1 + Im[i*3+2]*om2;
  float a[3];
  #pragma unroll
  for (int i = 0; i < 3; ++i)
    a[i] = DT * p[i] + DT * DT * (1.f - ALPHA) * M1[b*3 + i];

  float f[3] = {0.f, 0.f, 0.f};
  for (int it = 0; it < NSTEPS; ++it) {
    float aTf = a[0]*f[0] + a[1]*f[1] + a[2]*f[2];
    float cx0 = a[1]*f[2] - a[2]*f[1];
    float cx1 = a[2]*f[0] - a[0]*f[2];
    float cx2 = a[0]*f[1] - a[1]*f[0];
    float If0 = Im[0]*f[0] + Im[1]*f[1] + Im[2]*f[2];
    float If1 = Im[3]*f[0] + Im[4]*f[1] + Im[5]*f[2];
    float If2 = Im[6]*f[0] + Im[7]*f[1] + Im[8]*f[2];
    float phi[3] = { a[0] + cx0 + f[0]*aTf - 2.f*If0,
                     a[1] + cx1 + f[1]*aTf - 2.f*If1,
                     a[2] + cx2 + f[2]*aTf - 2.f*If2 };
    float D[9] = {  // hat(a) + aTf*I3 - 2*Im + outer(f,a)
      aTf  - 2.f*Im[0] + f[0]*a[0], -a[2] - 2.f*Im[1] + f[0]*a[1],  a[1] - 2.f*Im[2] + f[0]*a[2],
      a[2] - 2.f*Im[3] + f[1]*a[0],  aTf  - 2.f*Im[4] + f[1]*a[1], -a[0] - 2.f*Im[5] + f[1]*a[2],
     -a[1] - 2.f*Im[6] + f[2]*a[0],  a[0] - 2.f*Im[7] + f[2]*a[1],  aTf  - 2.f*Im[8] + f[2]*a[2]
    };
    float Di[9]; inv3(D, Di);
    #pragma unroll
    for (int i = 0; i < 3; ++i)
      f[i] -= Di[i*3]*phi[0] + Di[i*3+1]*phi[1] + Di[i*3+2]*phi[2];
  }

  // Cayley: F = (I + hat(f)) @ inv(I - hat(f))
  float P[9]  = { 1.f, -f[2],  f[1],   f[2], 1.f, -f[0],  -f[1],  f[0], 1.f };
  float Qm[9] = { 1.f,  f[2], -f[1],  -f[2], 1.f,  f[0],   f[1], -f[0], 1.f };
  float Qi[9]; inv3(Qm, Qi);
  float F[9];
  #pragma unroll
  for (int i = 0; i < 3; ++i)
    #pragma unroll
    for (int j = 0; j < 3; ++j)
      F[i*3+j] = P[i*3]*Qi[j] + P[i*3+1]*Qi[3+j] + P[i*3+2]*Qi[6+j];

  float Rm[9];
  #pragma unroll
  for (int i = 0; i < 9; ++i) Rm[i] = Rin[b*9 + i];
  #pragma unroll
  for (int i = 0; i < 3; ++i)
    #pragma unroll
    for (int j = 0; j < 3; ++j)
      Rnext[b*9 + i*3 + j] = Rm[i*3]*F[j] + Rm[i*3+1]*F[3+j] + Rm[i*3+2]*F[6+j];

  #pragma unroll
  for (int i = 0; i < 9; ++i) Fout[b*9 + i] = F[i];
  #pragma unroll
  for (int i = 0; i < 3; ++i) pout[b*3 + i] = p[i];
}

// ---------------------------------------------------------------------------
// K4: p_next = F^T (p + a*DT*M1) + (1-a)*DT*M2 ; omega_next = inv(I) p_next
// ---------------------------------------------------------------------------
__global__ void k_finish(const float* __restrict__ Iin, const float* __restrict__ M1,
                         const float* __restrict__ M2, const float* __restrict__ Fin,
                         const float* __restrict__ pin, float* __restrict__ om_out, int B)
{
  int b = blockIdx.x * blockDim.x + threadIdx.x;
  if (b >= B) return;
  float Im[9];
  #pragma unroll
  for (int i = 0; i < 9; ++i) Im[i] = Iin[i];
  float Ii[9]; inv3(Im, Ii);
  float F[9];
  #pragma unroll
  for (int i = 0; i < 9; ++i) F[i] = Fin[b*9 + i];
  float t[3];
  #pragma unroll
  for (int k = 0; k < 3; ++k) t[k] = pin[b*3 + k] + ALPHA * DT * M1[b*3 + k];
  float pn[3];
  #pragma unroll
  for (int i = 0; i < 3; ++i)
    pn[i] = F[0+i]*t[0] + F[3+i]*t[1] + F[6+i]*t[2] + (1.f - ALPHA) * DT * M2[b*3 + i];
  #pragma unroll
  for (int i = 0; i < 3; ++i)
    om_out[b*3 + i] = Ii[i*3]*pn[0] + Ii[i*3+1]*pn[1] + Ii[i*3+2]*pn[2];
}

// ---------------------------------------------------------------------------
extern "C" void kernel_launch(void* const* d_in, const int* in_sizes, int n_in,
                              void* d_out, int out_size, void* d_ws, size_t ws_size,
                              hipStream_t stream)
{
  (void)n_in; (void)out_size; (void)ws_size;
  const float* R  = (const float*)d_in[0];
  const float* om = (const float*)d_in[1];
  const float* I  = (const float*)d_in[2];
  const float* W1 = (const float*)d_in[3];
  const float* b1 = (const float*)d_in[4];
  const float* W2 = (const float*)d_in[5];
  const float* b2 = (const float*)d_in[6];
  const float* W3 = (const float*)d_in[7];
  const int B = in_sizes[0] / 9;

  char* ws = (char*)d_ws;
  __bf16* W2bf  = (__bf16*)ws;
  __bf16* W2tbf = (__bf16*)(ws + (size_t)HID * HID * 2);
  float*  M1 = (float*)(ws + 2 * (size_t)HID * HID * 2);
  float*  M2 = M1 + (size_t)B * 3;
  float*  Fb = M2 + (size_t)B * 3;
  float*  pb = Fb + (size_t)B * 9;

  float* Rnext  = (float*)d_out;
  float* om_out = Rnext + (size_t)B * 9;

  k_convert_w2<<<HID * HID / 256, 256, 0, stream>>>(W2, W2bf, W2tbf);
  k_explicitM<<<B / 16, 256, 0, stream>>>(R, W1, b1, W2bf, W2tbf, b2, W3, M1);
  k_newton<<<(B + 255) / 256, 256, 0, stream>>>(R, om, I, M1, Rnext, Fb, pb, B);
  k_explicitM<<<B / 16, 256, 0, stream>>>(Rnext, W1, b1, W2bf, W2tbf, b2, W3, M2);
  k_finish<<<(B + 255) / 256, 256, 0, stream>>>(I, M1, M2, Fb, pb, om_out, B);
}
